// MPNNLayer_60138132078771
// MI455X (gfx1250) — matrix operand order repaired and checked by reading, not compiled, not run on hardware
//
#include <hip/hip_runtime.h>
#include <hip/hip_bf16.h>

typedef __attribute__((ext_vector_type(16))) _Float16     v16h;
typedef __attribute__((ext_vector_type(8)))  _Float16     v8h;
typedef __attribute__((ext_vector_type(8)))  float        v8f;
typedef __attribute__((ext_vector_type(4)))  float        v4f;
typedef __attribute__((ext_vector_type(4)))  unsigned int v4u;
typedef __attribute__((ext_vector_type(8)))  int          v8i;
typedef __attribute__((ext_vector_type(4)))  int          v4i;

#define DIM 128
#define BN_EPS 1e-5f

// ---------------------------------------------------------------------------
// helpers
// ---------------------------------------------------------------------------
__device__ __forceinline__ v16h load_a16(const _Float16* base) {
    // A-operand layout (16-bit A 16x32): lane group g=lane>>4 reads
    // K = ks*32 + g*8 .. +7  (VGPR0-3) and K = ks*32 + 16 + g*8 .. +7 (VGPR4-7)
    v8h lo = *(const v8h*)(base);
    v8h hi = *(const v8h*)(base + 16);
    return __builtin_shufflevector(lo, hi, 0, 1, 2, 3, 4, 5, 6, 7,
                                           8, 9, 10, 11, 12, 13, 14, 15);
}

__device__ __forceinline__ v8f wmma16(v16h a, v16h b, v8f c) {
    return __builtin_amdgcn_wmma_f32_16x16x32_f16(
        /*neg_a=*/false, a, /*neg_b=*/false, b,
        /*c_mod=*/(short)0, c, /*reuse_a=*/false, /*reuse_b=*/false);
}

// ---------------------------------------------------------------------------
// K0a: aggr = x   (so edge scatter accumulates into x + sum(msg))
// ---------------------------------------------------------------------------
__global__ void __launch_bounds__(256) copy_x_kernel(const float* __restrict__ x,
                                                     float* __restrict__ aggr,
                                                     long n4) {
    long gid = (long)blockIdx.x * blockDim.x + threadIdx.x;
    if (gid < n4) ((v4f*)aggr)[gid] = ((const v4f*)x)[gid];
}

// ---------------------------------------------------------------------------
// K0b: pack a row-major f32 weight [K, Ncol] into WMMA B-operand f16 layout:
//   out[((ct*KS + ks)*32 + lane)*16 + j] = W[ks*32 + (lane>>4)*8 + (j&7) + (j>>3)*16][ct*16 + (lane&15)]
// One contiguous 32-byte load per lane per WMMA afterwards.
// ---------------------------------------------------------------------------
__global__ void __launch_bounds__(256) pack_b_kernel(const float* __restrict__ W,
                                                     _Float16* __restrict__ out,
                                                     int K, int Ncol) {
    int gid = blockIdx.x * blockDim.x + threadIdx.x;
    int total = K * Ncol;
    if (gid >= total) return;
    int j    = gid & 15;
    int lane = (gid >> 4) & 31;
    int blk  = gid >> 9;              // (ct * KS + ks)
    int KS   = K >> 5;
    int ks   = blk % KS;
    int ct   = blk / KS;
    int n    = ct * 16 + (lane & 15);
    int k    = ks * 32 + (lane >> 4) * 8 + (j & 7) + (j >> 3) * 16;
    out[gid] = (_Float16)W[k * Ncol + n];
}

// ---------------------------------------------------------------------------
// K1: edge messages: aggr[dst] += relu(x[src] + edge_attr[e])
// One thread per (edge, 4-float chunk). Scatter-adds hit L2 (51MB << 192MB).
// ---------------------------------------------------------------------------
__global__ void __launch_bounds__(256) edge_msg_kernel(const float* __restrict__ x,
                                                       const int*   __restrict__ ei,
                                                       const float* __restrict__ ea,
                                                       float* __restrict__ aggr,
                                                       int E) {
    long gid = (long)blockIdx.x * blockDim.x + threadIdx.x;
    int e = (int)(gid >> 5);                 // 32 chunks of float4 per 128-wide row
    if (e >= E) return;
    int c = ((int)gid & 31) << 2;
    int src = ei[e];
    int dst = ei[E + e];
    const float* eap = ea + (long)e * DIM + c;
    if (c == 0 && e + 256 < E)               // stream-ahead hint for edge_attr
        __builtin_prefetch(ea + ((long)e + 256) * DIM, 0, 1);
    v4f m = *(const v4f*)(x + (long)src * DIM + c) + *(const v4f*)eap;
    float* out = aggr + (long)dst * DIM + c;
#pragma unroll
    for (int j = 0; j < 4; ++j) {
        float v = m[j] > 0.f ? m[j] : 0.f;
        __hip_atomic_fetch_add(out + j, v, __ATOMIC_RELAXED, __HIP_MEMORY_SCOPE_AGENT);
    }
}

// ---------------------------------------------------------------------------
// K2: fused GIN node update:
//   h1 = BN1( x + ( relu(BN_gin(x + aggr)) @ gin_W + gin_b ) )
// 64-row tile per block (8 waves); wave w owns column tile w (16 cols),
// iterates 4 row tiles; K=128 -> 4 WMMA K-steps. A staged in LDS (f16).
// Also emits an f16 mirror of h1 for TDM staging in the FFN kernel.
// ---------------------------------------------------------------------------
__global__ void __launch_bounds__(256) gin_update_kernel(
    const float* __restrict__ hpre,     // x + sum(msg)
    const float* __restrict__ x,
    const float* __restrict__ gg, const float* __restrict__ gb,
    const float* __restrict__ gm, const float* __restrict__ gv,
    const _Float16* __restrict__ Wp,    // packed gin_W (B layout)
    const float* __restrict__ lb,       // gin_b
    const float* __restrict__ n1g, const float* __restrict__ n1b,
    const float* __restrict__ n1m, const float* __restrict__ n1v,
    float* __restrict__ h1, _Float16* __restrict__ h1h, int N) {
    __shared__ _Float16 lsA[64 * DIM];  // 16 KB
    int row0 = blockIdx.x * 64;
    int tid = threadIdx.x;

    // Phase A: A = f16(relu(BN_gin(hpre)))
    for (int t = tid; t < 64 * (DIM / 4); t += 256) {
        int r = t >> 5;
        int c = (t & 31) << 2;
        int row = row0 + r;
        if (row >= N) row = N - 1;
        v4f hv = *(const v4f*)(hpre + (long)row * DIM + c);
#pragma unroll
        for (int j = 0; j < 4; ++j) {
            float s = (hv[j] - gm[c + j]) * __frsqrt_rn(gv[c + j] + BN_EPS)
                      * gg[c + j] + gb[c + j];
            lsA[r * DIM + c + j] = (_Float16)(s > 0.f ? s : 0.f);
        }
    }
    __syncthreads();

    int wave = tid >> 5, lane = tid & 31;
    int hf = lane >> 4, ln = lane & 15;

    v8f acc[4] = {};
#pragma unroll
    for (int ks = 0; ks < 4; ++ks) {
        v16h bmat = *(const v16h*)(Wp + (((wave * 4 + ks) * 32 + lane) << 4));
#pragma unroll
        for (int rt = 0; rt < 4; ++rt) {
            v16h amat = load_a16(lsA + (rt * 16 + ln) * DIM + ks * 32 + hf * 8);
            acc[rt] = wmma16(amat, bmat, acc[rt]);
        }
    }

    // Epilogue: +gin_b, residual with x, BN1, store h1 (f32 + f16 mirror)
    int col = wave * 16 + ln;
    float gbias = lb[col];
    float im = n1m[col], iv = __frsqrt_rn(n1v[col] + BN_EPS);
    float ig = n1g[col], ib = n1b[col];
#pragma unroll
    for (int rt = 0; rt < 4; ++rt) {
#pragma unroll
        for (int r = 0; r < 8; ++r) {
            int row = row0 + rt * 16 + r + 8 * hf;   // C/D layout: VGPR r -> M=r+8*hf
            if (row < N) {
                float hval = acc[rt][r] + gbias;
                float s = x[(long)row * DIM + col] + hval;
                float o = (s - im) * iv * ig + ib;
                h1[(long)row * DIM + col] = o;
                h1h[(long)row * DIM + col] = (_Float16)o;
            }
        }
    }
}

// ---------------------------------------------------------------------------
// K3: fused FFN:
//   out = BN2( h1 + ( relu(h1 @ W1 + b1) @ W2 + b2 ) )
// A-tile (64x128 f16, contiguous 16KB) staged into LDS by the Tensor Data
// Mover (single descriptor, wave 0 issues, TENSORcnt-waited). Intermediate
// 64x256 lives in LDS as f16 (32 KB).
// ---------------------------------------------------------------------------
__global__ void __launch_bounds__(256) ffn_kernel(
    const float* __restrict__ h1,
    const _Float16* __restrict__ h1h,
    const _Float16* __restrict__ W1p, const float* __restrict__ b1,
    const _Float16* __restrict__ W2p, const float* __restrict__ b2,
    const float* __restrict__ n2g, const float* __restrict__ n2b,
    const float* __restrict__ n2m, const float* __restrict__ n2v,
    float* __restrict__ out, int N) {
    __shared__ _Float16 lsA[64 * DIM];       // 16 KB
    __shared__ _Float16 lsT[64 * 2 * DIM];   // 32 KB
    int row0 = blockIdx.x * 64;
    int tid = threadIdx.x;

    // ---- TDM: DMA h1h[row0*128 .. +8192) -> lsA (zero-fill past N) ----
    if (tid < 32) {  // one wave issues the tensor op (TDM ignores EXEC)
        unsigned long long ga =
            (unsigned long long)(const void*)(h1h + (long)row0 * DIM);
        unsigned int ldsOff = (unsigned int)(unsigned long long)(void*)lsA;
        unsigned int td0 = (unsigned int)N * DIM;     // tensor_dim0 (elements)

        v4u g0;
        g0[0] = 1u;                                   // count=1, user mode
        g0[1] = ldsOff;                               // lds_addr
        g0[2] = (unsigned int)ga;                     // global_addr[31:0]
        g0[3] = (unsigned int)((ga >> 32) & 0x1FFFFFFu) | (2u << 30); // +type=2

        v8i g1;
        g1[0] = (int)(1u << 16);                      // data_size=1 (2 bytes)
        g1[1] = (int)((td0 & 0xFFFFu) << 16);         // tensor_dim0[15:0]
        g1[2] = (int)(((td0 >> 16) & 0xFFFFu) | (1u << 16)); // dim0[31:16], tensor_dim1=1
        g1[3] = (int)(8192u << 16);                   // tile_dim0 = 64*128
        g1[4] = 0;                                    // tile_dim1/2 unused
        g1[5] = (int)td0;                             // tensor_dim0_stride lo
        g1[6] = 0;
        g1[7] = 0;

        v4i gz4 = {0, 0, 0, 0};
        v8i gz8 = {0, 0, 0, 0, 0, 0, 0, 0};
        __builtin_amdgcn_tensor_load_to_lds(g0, g1, gz4, gz4, gz8, 0);
        __builtin_amdgcn_s_wait_tensorcnt(0);
    }
    __syncthreads();

    int wave = tid >> 5, lane = tid & 31;
    int hf = lane >> 4, ln = lane & 15;

    // GEMM1 + ReLU -> lsT
#pragma unroll
    for (int cc = 0; cc < 2; ++cc) {
        int ct = wave + cc * 8;
        v8f acc[4] = {};
#pragma unroll
        for (int ks = 0; ks < 4; ++ks) {
            v16h bmat = *(const v16h*)(W1p + (((ct * 4 + ks) * 32 + lane) << 4));
#pragma unroll
            for (int rt = 0; rt < 4; ++rt) {
                v16h amat = load_a16(lsA + (rt * 16 + ln) * DIM + ks * 32 + hf * 8);
                acc[rt] = wmma16(amat, bmat, acc[rt]);
            }
        }
        int col = ct * 16 + ln;
        float bias = b1[col];
#pragma unroll
        for (int rt = 0; rt < 4; ++rt)
#pragma unroll
            for (int r = 0; r < 8; ++r) {
                float v = acc[rt][r] + bias;
                lsT[(rt * 16 + r + 8 * hf) * (2 * DIM) + col] =
                    (_Float16)(v > 0.f ? v : 0.f);
            }
    }
    __syncthreads();

    // GEMM2 + residual + BN2 -> out
    v8f acc[4] = {};
#pragma unroll
    for (int ks = 0; ks < 8; ++ks) {
        v16h bmat = *(const v16h*)(W2p + (((wave * 8 + ks) * 32 + lane) << 4));
#pragma unroll
        for (int rt = 0; rt < 4; ++rt) {
            v16h amat = load_a16(lsT + (rt * 16 + ln) * (2 * DIM) + ks * 32 + hf * 8);
            acc[rt] = wmma16(amat, bmat, acc[rt]);
        }
    }
    int col = wave * 16 + ln;
    float bias = b2[col];
    float im = n2m[col], iv = __frsqrt_rn(n2v[col] + BN_EPS);
    float ig = n2g[col], ib = n2b[col];
#pragma unroll
    for (int rt = 0; rt < 4; ++rt)
#pragma unroll
        for (int r = 0; r < 8; ++r) {
            int row = row0 + rt * 16 + r + 8 * hf;
            if (row < N) {
                float s = h1[(long)row * DIM + col] + acc[rt][r] + bias;
                out[(long)row * DIM + col] = (s - im) * iv * ig + ib;
            }
        }
}

// ---------------------------------------------------------------------------
// launch
// ---------------------------------------------------------------------------
extern "C" void kernel_launch(void* const* d_in, const int* in_sizes, int n_in,
                              void* d_out, int out_size, void* d_ws, size_t ws_size,
                              hipStream_t stream) {
    const float* x        = (const float*)d_in[0];
    const int*   ei       = (const int*)  d_in[1];
    const float* ea       = (const float*)d_in[2];
    const float* gin_bn_g = (const float*)d_in[3];
    const float* gin_bn_b = (const float*)d_in[4];
    const float* gin_bn_m = (const float*)d_in[5];
    const float* gin_bn_v = (const float*)d_in[6];
    const float* gin_W    = (const float*)d_in[7];
    const float* gin_b    = (const float*)d_in[8];
    const float* bn1_g    = (const float*)d_in[9];
    const float* bn1_b    = (const float*)d_in[10];
    const float* bn1_m    = (const float*)d_in[11];
    const float* bn1_v    = (const float*)d_in[12];
    const float* ffn_W1   = (const float*)d_in[13];
    const float* ffn_b1   = (const float*)d_in[14];
    const float* ffn_W2   = (const float*)d_in[15];
    const float* ffn_b2   = (const float*)d_in[16];
    const float* bn2_g    = (const float*)d_in[17];
    const float* bn2_b    = (const float*)d_in[18];
    const float* bn2_m    = (const float*)d_in[19];
    const float* bn2_v    = (const float*)d_in[20];

    int N = in_sizes[0] / DIM;
    int E = in_sizes[1] / 2;

    char* ws = (char*)d_ws;
    size_t nodeBytes = (size_t)N * DIM * sizeof(float);
    float*    aggr = (float*)ws;                          // [N,128] f32
    float*    h1   = (float*)(ws + nodeBytes);            // [N,128] f32
    _Float16* Wgp  = (_Float16*)(ws + 2 * nodeBytes);     // packed gin_W
    _Float16* W1p  = Wgp + DIM * DIM;                     // packed ffn_W1
    _Float16* W2p  = W1p + DIM * 2 * DIM;                 // packed ffn_W2
    _Float16* h1h  = W2p + 2 * DIM * DIM;                 // [N,128] f16 mirror

    // aggr = x  (accumulator starts at x so scatter produces x + sum(msg))
    long n4 = (long)N * DIM / 4;
    copy_x_kernel<<<(int)((n4 + 255) / 256), 256, 0, stream>>>(x, aggr, n4);

    // pack weights into WMMA B-operand f16 layout
    pack_b_kernel<<<(DIM * DIM + 255) / 256, 256, 0, stream>>>(gin_W, Wgp, DIM, DIM);
    pack_b_kernel<<<(DIM * 2 * DIM + 255) / 256, 256, 0, stream>>>(ffn_W1, W1p, DIM, 2 * DIM);
    pack_b_kernel<<<(2 * DIM * DIM + 255) / 256, 256, 0, stream>>>(ffn_W2, W2p, 2 * DIM, DIM);

    // edge messages (memory-bound stage)
    long et = (long)E * 32;
    edge_msg_kernel<<<(int)((et + 255) / 256), 256, 0, stream>>>(x, ei, ea, aggr, E);

    // fused node update + FFN (WMMA stages)
    int nwg = (N + 63) / 64;
    gin_update_kernel<<<nwg, 256, 0, stream>>>(aggr, x,
        gin_bn_g, gin_bn_b, gin_bn_m, gin_bn_v, Wgp, gin_b,
        bn1_g, bn1_b, bn1_m, bn1_v, h1, h1h, N);
    ffn_kernel<<<nwg, 256, 0, stream>>>(h1, h1h, W1p, ffn_b1, W2p, ffn_b2,
        bn2_g, bn2_b, bn2_m, bn2_v, (float*)d_out, N);
}
